// SIREN_vel_39307540693053
// MI455X (gfx1250) — matrix operand order repaired and checked by reading, not compiled, not run on hardware
//
#include <hip/hip_runtime.h>

typedef __attribute__((ext_vector_type(16))) _Float16 v16h;
typedef __attribute__((ext_vector_type(8)))  _Float16 v8h;
typedef __attribute__((ext_vector_type(8)))  float    v8f;
typedef __attribute__((ext_vector_type(4)))  float    v4f;

#define N_PTS   2097152
#define HID     128
#define NLAY    5                      // hidden WMMA layers
#define RPW     32                     // rows per wave (two 16-row M-tiles)
#define TILE    256                    // rows per block-iteration (8 waves x 32 rows)
#define ROWP    136                    // padded activation row pitch (f16 units)
#define WFRAG_N (NLAY * 8 * 4 * 32 * 16)   // 81920 f16 weight-fragment elements
#define ACT_N   (8 * RPW * ROWP)           // 34816 f16 activation elements
#define NBLK    1024
#define NTILES  (N_PTS / TILE)             // 8192

__launch_bounds__(256, 1)
__global__ void siren_wmma_kernel(const float* __restrict__ x,
                                  const float* __restrict__ W0,
                                  const float* __restrict__ b0,
                                  const float* __restrict__ Wh,
                                  const float* __restrict__ bh,
                                  const float* __restrict__ Wv,
                                  const float* __restrict__ bv,
                                  float* __restrict__ out)
{
    extern __shared__ __align__(32) char smem[];
    _Float16* wfrag = (_Float16*)smem;              // B-fragment-ordered f16 weights
    _Float16* act   = wfrag + WFRAG_N;              // per-wave activation tiles
    float* bh_l = (float*)(act + ACT_N);            // 5*128
    float* b0_l = bh_l + NLAY * HID;                // 128
    float* w0_l = b0_l + HID;                       // 128*4
    float* wv_l = w0_l + HID * 4;                   // 3*128
    float* bv_l = wv_l + 3 * HID;                   // 3

    const int tid  = threadIdx.x;
    const int lane = tid & 31;
    const int wid  = tid >> 5;        // wave 0..7
    const int half = lane >> 4;       // 0: lanes 0-15, 1: lanes 16-31
    const int l15  = lane & 15;

    // ---- one-time preload: convert weights to f16 in WMMA B-fragment order ----
    // flat i = (((L*8 + c)*4 + k)*32 + l)*16 + e
    // B 32x16 f16 layout: lane<16 holds K=0..7,16..23 (e), lane>=16 holds K=8..15,24..31
    for (int i = tid; i < WFRAG_N; i += 256) {
        int e = i & 15;
        int l = (i >> 4) & 31;
        int k = (i >> 9) & 3;
        int c = (i >> 11) & 7;
        int L = i >> 14;
        int n  = (c << 4) + (l & 15);                 // output column
        int kk = (k << 5) + ((l >> 4) << 3) + (e & 7) + ((e >> 3) << 4);      // K index
        wfrag[i] = (_Float16)Wh[(L * HID + n) * HID + kk];
    }
    for (int i = tid; i < NLAY * HID; i += 256) bh_l[i] = bh[i];
    for (int i = tid; i < HID;        i += 256) b0_l[i] = b0[i];
    for (int i = tid; i < HID * 4;    i += 256) w0_l[i] = W0[i];
    for (int i = tid; i < 3 * HID;    i += 256) wv_l[i] = Wv[i];
    if (tid < 3) bv_l[tid] = bv[tid];
    __syncthreads();

    _Float16* wact = act + wid * (RPW * ROWP);  // this wave's 32x128 activation tile

    for (int t = blockIdx.x; t < NTILES; t += gridDim.x) {
        const int r0 = t * TILE + wid * RPW;    // first point-row of this wave

        // ---------- first sine layer: h = sin(30*(x·W0^T + b0)), K=4 (VALU) ----------
        // lane -> row: all 32 lanes active; coalesced nontemporal x load (streamed once)
        {
            const v4f xv =
                __builtin_nontemporal_load(((const v4f*)x) + (r0 + lane));
            for (int cb = 0; cb < 16; ++cb) {
                v8h hv;
                #pragma unroll
                for (int e = 0; e < 8; ++e) {
                    const int c2 = cb * 8 + e;
                    float s = xv.x * w0_l[c2 * 4 + 0] + xv.y * w0_l[c2 * 4 + 1]
                            + xv.z * w0_l[c2 * 4 + 2] + xv.w * w0_l[c2 * 4 + 3]
                            + b0_l[c2];
                    hv[e] = (_Float16)__sinf(30.0f * s);
                }
                *(v8h*)(wact + lane * ROWP + cb * 8) = hv;   // ds_store_b128
            }
        }

        // ---------- 5 hidden layers: 32x128 @ 128x128 via v_wmma_f32_16x16x32_f16 ----
        // Two M-tiles per wave share every B fragment: halves B LDS traffic and
        // gives two independent accumulator chains per wave for XDL pipelining.
        for (int L = 0; L < NLAY; ++L) {
            // A fragments: 16-bit A 16x32 layout.
            // lane<16 row=l15: e0..7 = K 32k+0..7,  e8..15 = K 32k+16..23
            // lane>=16        : e0..7 = K 32k+8..15, e8..15 = K 32k+24..31
            v16h a0[4], a1[4];
            #pragma unroll
            for (int k = 0; k < 4; ++k) {
                const int o0 = (k << 5) + (half << 3);
                v8h lo0 = *(const v8h*)(wact + l15 * ROWP + o0);
                v8h hi0 = *(const v8h*)(wact + l15 * ROWP + o0 + 16);
                v8h lo1 = *(const v8h*)(wact + (16 + l15) * ROWP + o0);
                v8h hi1 = *(const v8h*)(wact + (16 + l15) * ROWP + o0 + 16);
                a0[k] = __builtin_shufflevector(lo0, hi0,
                        0, 1, 2, 3, 4, 5, 6, 7, 8, 9, 10, 11, 12, 13, 14, 15);
                a1[k] = __builtin_shufflevector(lo1, hi1,
                        0, 1, 2, 3, 4, 5, 6, 7, 8, 9, 10, 11, 12, 13, 14, 15);
            }
            const v16h* bf = (const v16h*)wfrag + (L * 8) * 4 * 32;
            #pragma unroll
            for (int c = 0; c < 8; ++c) {
                v8f acc0 = {};
                v8f acc1 = {};
                #pragma unroll
                for (int k = 0; k < 4; ++k) {
                    v16h b = bf[(c * 4 + k) * 32 + lane];
                    acc0 = __builtin_amdgcn_wmma_f32_16x16x32_f16(
                            false, a0[k], false, b, (short)0, acc0, false, false);
                    acc1 = __builtin_amdgcn_wmma_f32_16x16x32_f16(
                            false, a1[k], false, b, (short)0, acc1, false, false);
                }
                const float bias = bh_l[L * HID + (c << 4) + l15];
                // C layout: VGPR v -> row v (lanes 0-15) or v+8 (lanes 16-31); lane -> col
                #pragma unroll
                for (int v = 0; v < 8; ++v) {
                    float h0 = __sinf(acc0[v] + bias);
                    float h1 = __sinf(acc1[v] + bias);
                    wact[(v + (half << 3)) * ROWP + (c << 4) + l15]      = (_Float16)h0;
                    wact[(16 + v + (half << 3)) * ROWP + (c << 4) + l15] = (_Float16)h1;
                }
            }
        }

        // ---------- head: 3-wide linear layer (VALU), one row per lane ----------
        {
            #pragma unroll
            for (int c = 0; c < 3; ++c) {
                float acc2 = bv_l[c];
                #pragma unroll
                for (int jb = 0; jb < 16; ++jb) {
                    v8h hh = *(const v8h*)(wact + lane * ROWP + jb * 8);
                    #pragma unroll
                    for (int e = 0; e < 8; ++e)
                        acc2 += (float)hh[e] * wv_l[c * HID + jb * 8 + e];
                }
                __builtin_nontemporal_store(acc2,
                        out + (size_t)(r0 + lane) * 3 + c);
            }
        }
    }
}

extern "C" void kernel_launch(void* const* d_in, const int* in_sizes, int n_in,
                              void* d_out, int out_size, void* d_ws, size_t ws_size,
                              hipStream_t stream) {
    (void)in_sizes; (void)n_in; (void)d_ws; (void)ws_size; (void)out_size;
    const float* x  = (const float*)d_in[0];
    const float* W0 = (const float*)d_in[1];
    const float* b0 = (const float*)d_in[2];
    const float* Wh = (const float*)d_in[3];
    const float* bh = (const float*)d_in[4];
    const float* Wv = (const float*)d_in[5];
    const float* bv = (const float*)d_in[6];
    float* out = (float*)d_out;

    const size_t smem = (size_t)WFRAG_N * 2 + (size_t)ACT_N * 2
                      + (size_t)(NLAY * HID + HID + HID * 4 + 3 * HID + 4) * 4;
    (void)hipFuncSetAttribute((const void*)siren_wmma_kernel,
                              hipFuncAttributeMaxDynamicSharedMemorySize, (int)smem);
    siren_wmma_kernel<<<NBLK, 256, smem, stream>>>(x, W0, b0, Wh, bh, Wv, bv, out);
}
